// GRU_channel_48765058678883
// MI455X (gfx1250) — matrix operand order repaired and checked by reading, not compiled, not run on hardware
//
#include <hip/hip_runtime.h>
#include <hip/hip_bf16.h>

#define DEVINL __device__ __forceinline__

typedef __attribute__((ext_vector_type(16))) __bf16 v16bf;
typedef __attribute__((ext_vector_type(8)))  float  v8f;

union Frag16 {
  v16bf v;
  uint4 q[2];
};

DEVINL unsigned short f2bf(float f) {
  unsigned int u = __float_as_uint(f);
  u += 0x7fffu + ((u >> 16) & 1u);     // round-to-nearest-even
  return (unsigned short)(u >> 16);
}
DEVINL float bf2f(unsigned short s) { return __uint_as_float(((unsigned int)s) << 16); }
DEVINL float sigmoidf_(float x) { return 1.0f / (1.0f + __expf(-x)); }
DEVINL float relu_(float x) { return x > 0.f ? x : 0.f; }

// A-fragment (16x32 bf16, M x K): lanes 0-15 row M=lane, e0..7 -> K+0..7, e8..15 -> K+16..23
// lanes 16-31: e0..7 -> K+8..15, e8..15 -> K+24..31.  Caller passes p = &A[row*lda + kc + 8*hi].
DEVINL v16bf frag_a(const unsigned short* p) {
  Frag16 f;
  f.q[0] = *(const uint4*)(p);
  f.q[1] = *(const uint4*)(p + 16);
  return f.v;
}
// B-fragment (32x16 bf16, K x N): lane covers col n = lane&15, same K order as frag_a.
// Caller passes p = &W[n*K + kc + 8*hi] with W stored row-major (N x K), i.e. B(k,n) = W[n*K+k].
DEVINL v16bf frag_b(const unsigned short* p) {
  Frag16 f;
  f.q[0] = *(const uint4*)(p);
  f.q[1] = *(const uint4*)(p + 16);
  return f.v;
}
DEVINL v8f wmma_bf16(v16bf a, v16bf b, v8f c) {
  return __builtin_amdgcn_wmma_f32_16x16x32_bf16(false, a, false, b, (short)0, c, false, false);
}

// ---------------- front end: node projections + zero agg ----------------
__global__ __launch_bounds__(128)
void node_proj_kernel(const float* __restrict__ fa, const float* __restrict__ fb,
                      const float* __restrict__ fl,
                      const float* __restrict__ aw, const float* __restrict__ ab,
                      const float* __restrict__ bw, const float* __restrict__ bb,
                      const float* __restrict__ lw, const float* __restrict__ lb,
                      float* __restrict__ tree) {
  const long bn = blockIdx.x;          // b*64 + n
  const int j = threadIdx.x;           // output channel 0..127
  __shared__ float sA[39], sB[89], sC[18];
  if (j < 39) sA[j] = fa[bn * 39 + j];
  if (j < 89) sB[j] = fb[bn * 89 + j];
  if (j < 18) sC[j] = fl[bn * 18 + j];
  __syncthreads();
  float a = ab[j];
  for (int i = 0; i < 39; ++i) a += sA[i] * aw[i * 128 + j];
  float b = bb[j];
  for (int i = 0; i < 89; ++i) b += sB[i] * bw[i * 128 + j];
  float l = lb[j];
  for (int i = 0; i < 18; ++i) l += sC[i] * lw[i * 128 + j];
  float* row = tree + bn * 896;
  row[j]       = relu_(a);
  row[128 + j] = relu_(b);
  row[256 + j] = relu_(l);
  row[384 + j] = 0.f; row[512 + j] = 0.f; row[640 + j] = 0.f; row[768 + j] = 0.f;
}

// ---------------- edge messages: gather + segment-sum via atomics ----------------
__global__ __launch_bounds__(128)
void edge_msg_kernel(const float* __restrict__ ef, const int* __restrict__ esrc,
                     const int* __restrict__ edst, const float* __restrict__ ew,
                     const float* __restrict__ eb, float* __restrict__ tree) {
  const long be = blockIdx.x;          // b*128 + e
  const int j = threadIdx.x;
  const long b = be >> 7;
  __shared__ float sE[5];
  if (j < 5) sE[j] = ef[be * 5 + j];
  __syncthreads();
  float v = eb[j];
  for (int i = 0; i < 5; ++i) v += sE[i] * ew[i * 128 + j];
  v = relu_(v);
  const int s = esrc[be];
  const int d = edst[be];
  float* dst = tree + (b * 64 + s) * 896 + 384;      // agg region
  const float* nv = tree + (b * 64 + d) * 896;       // nodevec of dst
  atomicAdd(dst + j,       nv[j]);
  atomicAdd(dst + 128 + j, nv[128 + j]);
  atomicAdd(dst + 256 + j, nv[256 + j]);
  atomicAdd(dst + 384 + j, v);
}

// ---------------- converts ----------------
__global__ void cvt_bf16_kernel(const float* __restrict__ in, unsigned short* __restrict__ out, long n) {
  for (long i = blockIdx.x * (long)blockDim.x + threadIdx.x; i < n; i += (long)gridDim.x * blockDim.x)
    out[i] = f2bf(in[i]);
}
// att_w is (K=256, N=256) row-major; GEMM wants (N,K): out[n*256+k] = bf(in[k*256+n])
__global__ void cvt_t256_kernel(const float* __restrict__ in, unsigned short* __restrict__ out) {
  const int n = blockIdx.x, k = threadIdx.x;
  out[n * 256 + k] = f2bf(in[k * 256 + n]);
}

// ---------------- generic WMMA GEMM:  C = A(bf16, MxK) @ W^T + bias ----------------
// W stored row-major (Ncols x K).  Block = 4 waves, each wave 16x64 of C. Grid (M/64, Ncols/64).
__global__ __launch_bounds__(128)
void wmma_gemm_nt(const unsigned short* __restrict__ A, int lda,
                  const unsigned short* __restrict__ W, int K,
                  const float* __restrict__ bias,
                  float* __restrict__ C, int ldc) {
  const int lane = threadIdx.x & 31;
  const int wave = threadIdx.x >> 5;
  const int lo = lane & 15, hi = lane >> 4;
  const long m0 = (long)blockIdx.x * 64 + wave * 16;
  const int  n0 = blockIdx.y * 64;
  v8f acc[4] = {};
  const unsigned short* aRow = A + (m0 + lo) * (long)lda + 8 * hi;
  const unsigned short* wRow = W + (long)(n0 + lo) * K + 8 * hi;
  for (int kc = 0; kc < K; kc += 32) {
    __builtin_prefetch(aRow + kc + 128, 0, 1);       // global_prefetch hint for A chunk 4 k-steps ahead
    v16bf a = frag_a(aRow + kc);
#pragma unroll
    for (int nt = 0; nt < 4; ++nt) {
      v16bf b = frag_b(wRow + (long)nt * 16 * K + kc);
      acc[nt] = wmma_bf16(a, b, acc[nt]);
    }
  }
#pragma unroll
  for (int nt = 0; nt < 4; ++nt) {
    const int col = n0 + nt * 16 + lo;
    const float bv = bias ? bias[col] : 0.f;
#pragma unroll
    for (int r = 0; r < 8; ++r) {
      const long row = m0 + hi * 8 + r;
      C[row * (long)ldc + col] = acc[nt][r] + bv;
    }
  }
}

// ---------------- recurrent GRU scan (one wave = 16 batch rows, 64 steps) ----------------
// gi: (2, 32768, 384) precomputed x@wih.T + bih.   whh bf16 (2, 384, 128).  bhh (2, 384).
// out: (32768, 256) at column offset dir*128; bf16 (layer0 feeding next GEMM) or f32 (final vec).
template <bool BF16OUT>
__global__ __launch_bounds__(32)
void gru_scan_kernel(const float* __restrict__ gi, const unsigned short* __restrict__ whh,
                     const float* __restrict__ bhh, void* __restrict__ outp) {
  const int dir = blockIdx.y;
  const int lane = threadIdx.x & 31;
  const int lo = lane & 15, hi = lane >> 4;
  const long b0 = (long)blockIdx.x * 16;
  const float* giD = gi + (long)dir * 32768L * 384L;
  const unsigned short* whhD = whh + (long)dir * 384 * 128;
  const float* bhhD = bhh + dir * 384;

  __shared__ __align__(16) unsigned short hb[2][16 * 128];   // double-buffered bf16 hidden state
  for (int i = threadIdx.x; i < 16 * 128; i += 32) { hb[0][i] = 0; hb[1][i] = 0; }
  __syncthreads();

  for (int s = 0; s < 64; ++s) {
    const int t = dir ? (63 - s) : s;
    const int cur = s & 1, nxt = cur ^ 1;
#pragma unroll 1
    for (int j = 0; j < 8; ++j) {                    // 16 hidden cols per tile
      v8f ar = {}, az = {}, an = {};
#pragma unroll
      for (int kk = 0; kk < 4; ++kk) {               // K = 128
        const int kc = kk * 32;
        v16bf a = frag_a(&hb[cur][lo * 128 + kc + 8 * hi]);
        const unsigned short* wB = whhD + (long)(j * 16 + lo) * 128 + kc + 8 * hi;
        ar = wmma_bf16(a, frag_b(wB),             ar);
        az = wmma_bf16(a, frag_b(wB + 128 * 128), az);
        an = wmma_bf16(a, frag_b(wB + 256 * 128), an);
      }
      const int c = j * 16 + lo;
      const float br_ = bhhD[c], bz_ = bhhD[c + 128], bn_ = bhhD[c + 256];
#pragma unroll
      for (int r = 0; r < 8; ++r) {
        const int lr = hi * 8 + r;                   // local batch row 0..15
        const long grow = ((b0 + lr) * 64 + t) * 384L;
        const float rr = sigmoidf_(giD[grow + c]       + ar[r] + br_);
        const float zz = sigmoidf_(giD[grow + c + 128] + az[r] + bz_);
        const float nn = tanhf   (giD[grow + c + 256] + rr * (an[r] + bn_));
        const float hn = (1.f - zz) * nn + zz * bf2f(hb[cur][lr * 128 + c]);
        hb[nxt][lr * 128 + c] = f2bf(hn);
        const long orow = ((b0 + lr) * 64 + t) * 256L + dir * 128 + c;
        if (BF16OUT) ((unsigned short*)outp)[orow] = f2bf(hn);
        else         ((float*)outp)[orow] = hn;
      }
    }
    __syncthreads();
  }
}

// ---------------- attention pooling epilogue (per batch element) ----------------
__global__ __launch_bounds__(256)
void attn_kernel(const float* __restrict__ vec, const float* __restrict__ Wv,
                 const float* __restrict__ att_w, const float* __restrict__ att_b,
                 const float* __restrict__ a_w, const float* __restrict__ a_b,
                 float* __restrict__ outp) {
  const long b = blockIdx.x;
  const int tid = threadIdx.x;                       // 256 threads = output channels
  __shared__ float sMean[256], red[256], sAlpha[64];
  __shared__ float sWsDot, sSum;
  float m = 0.f;
  for (int t = 0; t < 64; ++t) m += vec[(b * 64 + t) * 256 + tid];
  sMean[tid] = m * (1.f / 64.f);
  __syncthreads();
  float ws = att_b[tid];
  for (int k = 0; k < 256; ++k) ws += sMean[k] * att_w[k * 256 + tid];
  red[tid] = ws * a_w[tid];
  __syncthreads();
  for (int o = 128; o > 0; o >>= 1) {                // dot(Ws, a_w[0:256])
    if (tid < o) red[tid] += red[tid + o];
    __syncthreads();
  }
  if (tid == 0) sWsDot = red[0];
  __syncthreads();
  const int n = tid >> 2, q = tid & 3;               // 4 threads per node score
  float p = 0.f;
  const float* wvRow = Wv + (b * 64 + n) * 256;
  for (int k = q * 64; k < q * 64 + 64; ++k) p += wvRow[k] * a_w[256 + k];
  red[tid] = p;
  __syncthreads();
  if (q == 0) {
    float sc = red[tid] + red[tid + 1] + red[tid + 2] + red[tid + 3] + sWsDot + a_b[0];
    sc = sc > 0.f ? sc : 0.01f * sc;                 // leaky_relu(0.01)
    sAlpha[n] = __expf(sc);
  }
  __syncthreads();
  if (tid == 0) {
    float s = 0.f;
    for (int i = 0; i < 64; ++i) s += sAlpha[i];
    sSum = s;
  }
  __syncthreads();
  float acc = 0.f;
  for (int i = 0; i < 64; ++i) acc += sAlpha[i] * Wv[(b * 64 + i) * 256 + tid];
  outp[b * 256 + tid] = relu_(acc / sSum);
}

extern "C" void kernel_launch(void* const* d_in, const int* in_sizes, int n_in,
                              void* d_out, int out_size, void* d_ws, size_t ws_size,
                              hipStream_t stream) {
  (void)in_sizes; (void)n_in; (void)out_size; (void)ws_size;
  const float* fa        = (const float*)d_in[0];
  const float* fb        = (const float*)d_in[1];
  const float* fl        = (const float*)d_in[2];
  const float* edge_feat = (const float*)d_in[3];
  const int*   edge_src  = (const int*)d_in[4];
  const int*   edge_dst  = (const int*)d_in[5];
  const float* atom_w = (const float*)d_in[6];
  const float* atom_b = (const float*)d_in[7];
  const float* bond_w = (const float*)d_in[8];
  const float* bond_b = (const float*)d_in[9];
  const float* link_w = (const float*)d_in[10];
  const float* link_b = (const float*)d_in[11];
  const float* edgew_w = (const float*)d_in[12];
  const float* edgew_b = (const float*)d_in[13];
  const float* gru0_wih = (const float*)d_in[14];
  const float* gru0_whh = (const float*)d_in[15];
  const float* gru0_bih = (const float*)d_in[16];
  const float* gru0_bhh = (const float*)d_in[17];
  const float* gru1_wih = (const float*)d_in[18];
  const float* gru1_whh = (const float*)d_in[19];
  const float* gru1_bih = (const float*)d_in[20];
  const float* gru1_bhh = (const float*)d_in[21];
  const float* att_w = (const float*)d_in[22];
  const float* att_b = (const float*)d_in[23];
  const float* a_w   = (const float*)d_in[24];
  const float* a_b   = (const float*)d_in[25];

  const long BN = 512L * 64;                         // 32768 rows
  char* base = (char*)d_ws;
  size_t off = 0;
  auto take = [&](size_t bytes) {
    char* p = base + off;
    off += (bytes + 255) & ~(size_t)255;
    return p;
  };
  unsigned short* wih0bf = (unsigned short*)take(2L * 384 * 896 * 2);
  unsigned short* whh0bf = (unsigned short*)take(2L * 384 * 128 * 2);
  unsigned short* wih1bf = (unsigned short*)take(2L * 384 * 256 * 2);
  unsigned short* whh1bf = (unsigned short*)take(2L * 384 * 128 * 2);
  unsigned short* attwT  = (unsigned short*)take(256L * 256 * 2);
  char* regA = take(BN * 896 * 4);                   // tree (f32)  -> gi1
  char* regB = take(BN * 896 * 2);                   // treebf      -> h1bf
  char* regC = take(2L * BN * 384 * 4);              // gi0         -> {vec, vecbf, Wv}
  float*          tree   = (float*)regA;
  float*          gi1    = (float*)regA;
  unsigned short* treebf = (unsigned short*)regB;
  unsigned short* h1bf   = (unsigned short*)regB;
  float*          gi0    = (float*)regC;
  float*          vec    = (float*)regC;
  unsigned short* vecbf  = (unsigned short*)(regC + BN * 256 * 4);
  float*          Wv     = (float*)(regC + BN * 256 * 4 + BN * 256 * 2);

  dim3 cvtGrid(2048), cvtBlk(256);
  cvt_bf16_kernel<<<cvtGrid, cvtBlk, 0, stream>>>(gru0_wih, wih0bf, 2L * 384 * 896);
  cvt_bf16_kernel<<<cvtGrid, cvtBlk, 0, stream>>>(gru0_whh, whh0bf, 2L * 384 * 128);
  cvt_bf16_kernel<<<cvtGrid, cvtBlk, 0, stream>>>(gru1_wih, wih1bf, 2L * 384 * 256);
  cvt_bf16_kernel<<<cvtGrid, cvtBlk, 0, stream>>>(gru1_whh, whh1bf, 2L * 384 * 128);
  cvt_t256_kernel<<<256, 256, 0, stream>>>(att_w, attwT);

  node_proj_kernel<<<dim3((unsigned)BN), 128, 0, stream>>>(fa, fb, fl, atom_w, atom_b,
                                                           bond_w, bond_b, link_w, link_b, tree);
  edge_msg_kernel<<<dim3(512 * 128), 128, 0, stream>>>(edge_feat, edge_src, edge_dst,
                                                       edgew_w, edgew_b, tree);
  cvt_bf16_kernel<<<cvtGrid, cvtBlk, 0, stream>>>(tree, treebf, BN * 896);

  for (int d = 0; d < 2; ++d)
    wmma_gemm_nt<<<dim3(512, 6), 128, 0, stream>>>(treebf, 896, wih0bf + (long)d * 384 * 896, 896,
                                                   gru0_bih + d * 384, gi0 + (long)d * BN * 384, 384);
  gru_scan_kernel<true><<<dim3(32, 2), 32, 0, stream>>>(gi0, whh0bf, gru0_bhh, h1bf);

  for (int d = 0; d < 2; ++d)
    wmma_gemm_nt<<<dim3(512, 6), 128, 0, stream>>>(h1bf, 256, wih1bf + (long)d * 384 * 256, 256,
                                                   gru1_bih + d * 384, gi1 + (long)d * BN * 384, 384);
  gru_scan_kernel<false><<<dim3(32, 2), 32, 0, stream>>>(gi1, whh1bf, gru1_bhh, vec);

  cvt_bf16_kernel<<<cvtGrid, cvtBlk, 0, stream>>>(vec, vecbf, BN * 256);
  wmma_gemm_nt<<<dim3(512, 4), 128, 0, stream>>>(vecbf, 256, attwT, 256, att_b, Wv, 256);
  attn_kernel<<<512, 256, 0, stream>>>(vec, Wv, att_w, att_b, a_w, a_b, (float*)d_out);
}
